// MyRNN_40587440947604
// MI455X (gfx1250) — compile-verified, matrix operand order checked by reading, not run on hardware
//
#include <hip/hip_runtime.h>
#include <hip/hip_bf16.h>
#include <math.h>

// ---------------------------------------------------------------------------
// Types for CDNA5 WMMA (wave32): v_wmma_f32_16x16x32_bf16
// ---------------------------------------------------------------------------
typedef __bf16 bf16_t;
typedef __attribute__((ext_vector_type(16))) __bf16 v16bf;
typedef __attribute__((ext_vector_type(8)))  __bf16 v8bf;
typedef __attribute__((ext_vector_type(8)))  float  v8f;

// Problem dims
#define TT 512
#define BB 128
#define II 256
#define HH 512
#define OO 256

// LDS row stride for staged weight slices: 512 + 8 halves -> 260 words/row,
// 260 % 64 == 4 so the 16 lanes of a half-group stripe all 64 banks.
#define LDSTR 520

__device__ __forceinline__ v8f wmma_bf16(v16bf a, v16bf b, v8f c) {
  // (neg_a, A, neg_b, B, c_mod, C, reuse_a, reuse_b)
  return __builtin_amdgcn_wmma_f32_16x16x32_bf16(false, a, false, b,
                                                 (short)0, c, false, false);
}

// Load one 16x32 bf16 operand tile from global (A layout; B mirrored over N,
// so loading W rows n over k with the same addressing yields B == W^T (KxN)).
// Lane l: row = base_row + (l&15); VGPR0-3 hold K = k0 + (l>>4)*8 .. +7,
// VGPR4-7 hold K = k0 + 16 + (l>>4)*8 .. +7  (matches ISA 16-bit A table).
__device__ __forceinline__ v16bf load_tile16(const bf16_t* __restrict__ base,
                                             int row_base, int ld, int k0,
                                             int lane) {
  const int r = lane & 15;
  const int half = lane >> 4;
  const bf16_t* p = base + (size_t)(row_base + r) * (size_t)ld + k0 + half * 8;
  v8bf lo = *(const v8bf*)p;         // K = k0 + half*8 + [0..7]
  v8bf hi = *(const v8bf*)(p + 16);  // K = k0 + 16 + half*8 + [0..7]
  return __builtin_shufflevector(lo, hi, 0, 1, 2, 3, 4, 5, 6, 7,
                                 8, 9, 10, 11, 12, 13, 14, 15);
}

// Low 32 bits of a flat shared-memory pointer == LDS byte offset (ISA 10.2).
__device__ __forceinline__ unsigned lds_off_u32(const void* p) {
  return (unsigned)(uintptr_t)p;
}

// ---------------------------------------------------------------------------
// Utility kernels
// ---------------------------------------------------------------------------
__global__ void k_zero_u32(unsigned* __restrict__ p, int n) {
  for (int i = blockIdx.x * blockDim.x + threadIdx.x; i < n;
       i += gridDim.x * blockDim.x)
    p[i] = 0u;
}

__global__ void k_cvt_bf16(const float* __restrict__ src,
                           bf16_t* __restrict__ dst, int n) {
  for (int i = blockIdx.x * blockDim.x + threadIdx.x; i < n;
       i += gridDim.x * blockDim.x)
    dst[i] = (bf16_t)src[i];
}

// ---------------------------------------------------------------------------
// Phase 1: Z0 = bf16( x @ W_ih0^T + b_ih0 + b_hh0 )   [T*B, H]
// ---------------------------------------------------------------------------
__global__ __launch_bounds__(256) void k_z0_gemm(
    const bf16_t* __restrict__ xbf,     // [T*B, I]
    const bf16_t* __restrict__ Wih0,    // [H, I]
    const float* __restrict__ b_ih0, const float* __restrict__ b_hh0,
    bf16_t* __restrict__ Z0) {          // [T*B, H]
  const int lane = threadIdx.x & 31;
  const int wgid = blockIdx.x * 8 + (threadIdx.x >> 5);  // 0..131071
  const int n_base = (wgid & 31) * 16;
  const int m_base = (wgid >> 5) * 16;
  const int half = lane >> 4;
  const int n = n_base + (lane & 15);

  const float bias = b_ih0[n] + b_hh0[n];
  v8f acc;
#pragma unroll
  for (int i = 0; i < 8; ++i) acc[i] = bias;

#pragma unroll
  for (int k0 = 0; k0 < II; k0 += 32) {
    v16bf a = load_tile16(xbf, m_base, II, k0, lane);
    v16bf b = load_tile16(Wih0, n_base, II, k0, lane);
    acc = wmma_bf16(a, b, acc);
  }
#pragma unroll
  for (int i = 0; i < 8; ++i) {
    int m = m_base + i + half * 8;
    Z0[(size_t)m * HH + n] = (bf16_t)acc[i];
  }
}

// ---------------------------------------------------------------------------
// Phase 2: persistent 512-step recurrence. 32 blocks x 256 threads = 256 waves,
// one 16x16 tile per wave per GEMM phase. Block b owns N-columns [16b,16b+16):
// its three weight slices (16 rows x 512 each) are staged once into LDS via
// async global->LDS loads, then every step reads B operands with ds_load_b128.
// ---------------------------------------------------------------------------
#define RNN_BLOCKS 32

__device__ __forceinline__ void grid_sync(unsigned* __restrict__ bar,
                                          unsigned target) {
  __threadfence();   // release this wave's stores to device scope
  __syncthreads();
  if (threadIdx.x == 0) {
    __hip_atomic_fetch_add(bar, 1u, __ATOMIC_ACQ_REL, __HIP_MEMORY_SCOPE_AGENT);
    while (__hip_atomic_load(bar, __ATOMIC_ACQUIRE, __HIP_MEMORY_SCOPE_AGENT) <
           target)
      __builtin_amdgcn_s_sleep(2);
  }
  __syncthreads();
  __threadfence();   // acquire side
}

__global__ __launch_bounds__(256) void k_rnn_steps(
    const bf16_t* __restrict__ Z0,     // [T*B, H]
    bf16_t* __restrict__ h1,           // 2 x [B, H] (double buffer)
    bf16_t* __restrict__ h2,           // 2 x [B, H]
    const bf16_t* __restrict__ Whh0,   // [H, H]
    const bf16_t* __restrict__ Wih1,   // [H, H]
    const bf16_t* __restrict__ Whh1,   // [H, H]
    const float* __restrict__ b_ih1, const float* __restrict__ b_hh1,
    bf16_t* __restrict__ Ylast,        // [T, H]  (h2 row 127 per step)
    float* __restrict__ hfin,          // [2, B, H] fp32 (d_out tail)
    unsigned* __restrict__ bar) {
  __shared__ __align__(16) bf16_t smem[3 * 16 * LDSTR];

  const int lane = threadIdx.x & 31;
  const int wgid = blockIdx.x * 8 + (threadIdx.x >> 5);  // 0..255
  const int m_base = (wgid & 7) * 16;
  const int n_base = (wgid >> 3) * 16;   // == blockIdx.x * 16
  const int half = lane >> 4;
  const int n = n_base + (lane & 15);

  // ---- one-time async staging of this block's three weight slices ----
  {
#pragma unroll
    for (int mat = 0; mat < 3; ++mat) {
      const bf16_t* wsrc = (mat == 0) ? Whh0 : (mat == 1) ? Wih1 : Whh1;
      for (int c = threadIdx.x; c < 16 * 64; c += 256) {  // 64 x 16B per row
        int row = c >> 6;
        int c16 = c & 63;
        unsigned lds =
            lds_off_u32(&smem[(mat * 16 + row) * LDSTR + c16 * 8]);
        const bf16_t* g = wsrc + (size_t)(n_base + row) * HH + c16 * 8;
        asm volatile("global_load_async_to_lds_b128 %0, %1, off"
                     :: "v"(lds), "v"((unsigned long long)(uintptr_t)g)
                     : "memory");
      }
    }
    asm volatile("s_wait_asynccnt 0x0" ::: "memory");
    __syncthreads();
  }

  // B-operand tile from the staged LDS slice (same per-lane layout as A).
  auto ldsB = [&](int mat, int k0) -> v16bf {
    const bf16_t* p =
        &smem[(mat * 16 + (lane & 15)) * LDSTR + k0 + half * 8];
    v8bf lo = *(const v8bf*)p;
    v8bf hi = *(const v8bf*)(p + 16);
    return __builtin_shufflevector(lo, hi, 0, 1, 2, 3, 4, 5, 6, 7,
                                   8, 9, 10, 11, 12, 13, 14, 15);
  };

  const float bias2 = b_ih1[n] + b_hh1[n];
  unsigned epoch = 0;

  for (int t = 0; t < TT; ++t) {
    const bf16_t* h1r = h1 + ((t + 1) & 1) * (BB * HH);
    bf16_t* h1w = h1 + (t & 1) * (BB * HH);
    const bf16_t* h2r = h2 + ((t + 1) & 1) * (BB * HH);
    bf16_t* h2w = h2 + (t & 1) * (BB * HH);
    const bf16_t* z = Z0 + (size_t)t * (BB * HH);

    // Prefetch next step's Z0 tile rows into cache while we compute.
    if (t + 1 < TT) {
      const bf16_t* zn = Z0 + (size_t)(t + 1) * (BB * HH) +
                         (size_t)(m_base + (lane & 15)) * HH + n_base;
      __builtin_prefetch(zn, 0, 0);
    }

    // ---- layer 1: h1 = tanh(Z0[t] + h1_prev @ Whh0^T) ----
    v8f acc;
#pragma unroll
    for (int i = 0; i < 8; ++i)
      acc[i] = (float)z[(size_t)(m_base + i + half * 8) * HH + n];
#pragma unroll 4
    for (int k0 = 0; k0 < HH; k0 += 32) {
      v16bf a = load_tile16(h1r, m_base, HH, k0, lane);
      acc = wmma_bf16(a, ldsB(0, k0), acc);
    }
#pragma unroll
    for (int i = 0; i < 8; ++i) {
      int m = m_base + i + half * 8;
      float v = tanhf(acc[i]);
      h1w[(size_t)m * HH + n] = (bf16_t)v;
      if (t == TT - 1) hfin[(size_t)m * HH + n] = v;
    }
    grid_sync(bar, (++epoch) * RNN_BLOCKS);

    // ---- layer 2: h2 = tanh(h1 @ Wih1^T + h2_prev @ Whh1^T + bias2) ----
    v8f acc2;
#pragma unroll
    for (int i = 0; i < 8; ++i) acc2[i] = bias2;
#pragma unroll 4
    for (int k0 = 0; k0 < HH; k0 += 32) {
      v16bf a = load_tile16(h1w, m_base, HH, k0, lane);
      acc2 = wmma_bf16(a, ldsB(1, k0), acc2);
    }
#pragma unroll 4
    for (int k0 = 0; k0 < HH; k0 += 32) {
      v16bf a = load_tile16(h2r, m_base, HH, k0, lane);
      acc2 = wmma_bf16(a, ldsB(2, k0), acc2);
    }
#pragma unroll
    for (int i = 0; i < 8; ++i) {
      int m = m_base + i + half * 8;
      float v = tanhf(acc2[i]);
      h2w[(size_t)m * HH + n] = (bf16_t)v;
      if (m == BB - 1) Ylast[(size_t)t * HH + n] = (bf16_t)v;
      if (t == TT - 1) hfin[(size_t)(BB * HH) + (size_t)m * HH + n] = v;
    }
    grid_sync(bar, (++epoch) * RNN_BLOCKS);
  }
}

// ---------------------------------------------------------------------------
// Phase 3: pred = Ylast @ W_fc^T + b_fc   [T, O] = [512,512] x [512,256]
// ---------------------------------------------------------------------------
__global__ __launch_bounds__(256) void k_fc_gemm(
    const bf16_t* __restrict__ Ylast,   // [T, H]
    const bf16_t* __restrict__ Wfc,     // [O, H]
    const float* __restrict__ b_fc,
    float* __restrict__ pred) {         // [T, O]
  const int lane = threadIdx.x & 31;
  const int wgid = blockIdx.x * 8 + (threadIdx.x >> 5);  // 0..511
  const int n_base = (wgid & 15) * 16;
  const int m_base = (wgid >> 4) * 16;
  const int half = lane >> 4;
  const int n = n_base + (lane & 15);

  const float bias = b_fc[n];
  v8f acc;
#pragma unroll
  for (int i = 0; i < 8; ++i) acc[i] = bias;

#pragma unroll 4
  for (int k0 = 0; k0 < HH; k0 += 32) {
    v16bf a = load_tile16(Ylast, m_base, HH, k0, lane);
    v16bf b = load_tile16(Wfc, n_base, HH, k0, lane);
    acc = wmma_bf16(a, b, acc);
  }
#pragma unroll
  for (int i = 0; i < 8; ++i) {
    int m = m_base + i + half * 8;
    pred[(size_t)m * OO + n] = acc[i];
  }
}

// ---------------------------------------------------------------------------
// Launcher
// ---------------------------------------------------------------------------
extern "C" void kernel_launch(void* const* d_in, const int* in_sizes, int n_in,
                              void* d_out, int out_size, void* d_ws,
                              size_t ws_size, hipStream_t stream) {
  (void)in_sizes; (void)n_in; (void)out_size; (void)ws_size;

  const float* x     = (const float*)d_in[0];   // [T,B,I]
  const float* W_ih0 = (const float*)d_in[1];   // [H,I]
  const float* b_ih0 = (const float*)d_in[2];
  const float* W_hh0 = (const float*)d_in[3];   // [H,H]
  const float* b_hh0 = (const float*)d_in[4];
  const float* W_ih1 = (const float*)d_in[5];   // [H,H]
  const float* b_ih1 = (const float*)d_in[6];
  const float* W_hh1 = (const float*)d_in[7];   // [H,H]
  const float* b_hh1 = (const float*)d_in[8];
  const float* W_fc  = (const float*)d_in[9];   // [O,H]
  const float* b_fc  = (const float*)d_in[10];

  float* pred = (float*)d_out;                  // [T,O] = 131072 floats
  float* hfin = (float*)d_out + TT * OO;        // [2,B,H] = 131072 floats

  // Workspace carve-up (256B aligned)
  char* ws = (char*)d_ws;
  size_t off = 0;
  auto carve = [&](size_t bytes) -> char* {
    char* p = ws + off;
    off += (bytes + 255) & ~(size_t)255;
    return p;
  };
  unsigned* bar  = (unsigned*)carve(256);
  bf16_t* h1bf   = (bf16_t*)carve((size_t)2 * BB * HH * 2);
  bf16_t* h2bf   = (bf16_t*)carve((size_t)2 * BB * HH * 2);
  size_t zero_bytes = off;  // barrier + both state double-buffers
  bf16_t* Ylast  = (bf16_t*)carve((size_t)TT * HH * 2);
  bf16_t* xbf    = (bf16_t*)carve((size_t)TT * BB * II * 2);
  bf16_t* Wih0bf = (bf16_t*)carve((size_t)HH * II * 2);
  bf16_t* Whh0bf = (bf16_t*)carve((size_t)HH * HH * 2);
  bf16_t* Wih1bf = (bf16_t*)carve((size_t)HH * HH * 2);
  bf16_t* Whh1bf = (bf16_t*)carve((size_t)HH * HH * 2);
  bf16_t* Wfcbf  = (bf16_t*)carve((size_t)OO * HH * 2);
  bf16_t* Z0bf   = (bf16_t*)carve((size_t)TT * BB * HH * 2);

  // 0) deterministic re-init of barrier + hidden-state buffers every call
  {
    int n = (int)(zero_bytes / 4);
    k_zero_u32<<<128, 256, 0, stream>>>((unsigned*)ws, n);
  }

  // 1) fp32 -> bf16 staging of x and weights
  auto cvt = [&](const float* s, bf16_t* d, int n) {
    int blocks = (n + 255) / 256;
    if (blocks > 4096) blocks = 4096;
    k_cvt_bf16<<<blocks, 256, 0, stream>>>(s, d, n);
  };
  cvt(x,     xbf,    TT * BB * II);
  cvt(W_ih0, Wih0bf, HH * II);
  cvt(W_hh0, Whh0bf, HH * HH);
  cvt(W_ih1, Wih1bf, HH * HH);
  cvt(W_hh1, Whh1bf, HH * HH);
  cvt(W_fc,  Wfcbf,  OO * HH);

  // 2) recurrence-free input projection for all timesteps (one big WMMA GEMM)
  k_z0_gemm<<<16384, 256, 0, stream>>>(xbf, Wih0bf, b_ih0, b_hh0, Z0bf);

  // 3) persistent 512-step recurrence (latency-critical serial chain)
  k_rnn_steps<<<RNN_BLOCKS, 256, 0, stream>>>(Z0bf, h1bf, h2bf, Whh0bf, Wih1bf,
                                              Whh1bf, b_ih1, b_hh1, Ylast,
                                              hfin, bar);

  // 4) final FC on per-step last-batch rows
  k_fc_gemm<<<64, 256, 0, stream>>>(Ylast, Wfcbf, b_fc, pred);
}
